// UnifiedZipAdapterF_73744588472842
// MI455X (gfx1250) — matrix operand order repaired and checked by reading, not compiled, not run on hardware
//
#include <hip/hip_runtime.h>
#include <hip/hip_bf16.h>
#include <math.h>

// ---------------------------------------------------------------------------
// UnifiedZipAdapter pipeline for MI455X (gfx1250), bf16 WMMA path.
// Round 2: pre-converted bf16 operands for W / prototypes / embeddings so the
// GEMM inner loops are pure {global_load_b128 (bf16) -> v_wmma} with no
// per-iteration cvt VALU work and no WMMA->VALU hazard NOPs.
// ---------------------------------------------------------------------------

typedef __attribute__((ext_vector_type(16))) __bf16 v16bf;
typedef __attribute__((ext_vector_type(8)))  __bf16 v8bf;
typedef __attribute__((ext_vector_type(8)))  float  v8f;

#define DD     768
#define CC     1000
#define TAU_   0.11f
#define ALPHA_ 32.0f
#define EPS_   1e-12f

__device__ __forceinline__ v8f wmma_bf16(v16bf a, v16bf b, v8f c) {
  // D = A(16x32 bf16) * B(32x16 bf16) + C(16x16 f32)
  return __builtin_amdgcn_wmma_f32_16x16x32_bf16(
      /*neg_a=*/false, a, /*neg_b=*/false, b,
      /*c_mod=*/(short)0, c, /*reuse_a=*/false, /*reuse_b=*/false);
}

__device__ __forceinline__ v16bf join16(v8bf lo, v8bf hi) {
  return __builtin_shufflevector(lo, hi, 0,1,2,3,4,5,6,7,8,9,10,11,12,13,14,15);
}

__device__ __forceinline__ v16bf zero16bf() {
  v16bf r;
#pragma unroll
  for (int i = 0; i < 16; ++i) r[i] = (__bf16)0.0f;
  return r;
}

// ---------------------------------------------------------------------------
// cvt_bf16_kernel: elementwise f32 -> bf16 (for W and for embeddings copy).
// n must be a multiple of 4 (true for 768*768 and 8192*768).
// ---------------------------------------------------------------------------
__global__ __launch_bounds__(256)
void cvt_bf16_kernel(const float* __restrict__ in, __bf16* __restrict__ out, int n4)
{
  int i = blockIdx.x * blockDim.x + threadIdx.x;
  if (i < n4) {
    float4 v = *(const float4*)(in + (size_t)i * 4);
    v8bf dummy; (void)dummy;
    __bf16* o = out + (size_t)i * 4;
    o[0] = (__bf16)v.x; o[1] = (__bf16)v.y; o[2] = (__bf16)v.z; o[3] = (__bf16)v.w;
  }
}

// ---------------------------------------------------------------------------
// adapter_kernel: H = silu(T @ W^T + b) + T, row-L2-normalized.
//   mode 0: write normalized rows to out_emb (f32)
//   mode 1: atomic segment-sum into sums[labels[row]], counts[labels[row]]
// Block: 256 thr (8 waves). Tile: 32 rows x 768 cols. K loop: 24 x 32.
// W is pre-converted bf16 (row-major), resident in L2.
// ---------------------------------------------------------------------------
__global__ __launch_bounds__(256)
void adapter_kernel(const float* __restrict__ T, const __bf16* __restrict__ Wbf,
                    const float* __restrict__ bias, const int* __restrict__ labels,
                    int R, float* __restrict__ out_emb,
                    float* __restrict__ sums, float* __restrict__ counts, int mode)
{
  __shared__ __bf16 sA[32 * DD];   // 48 KB: 32 staged rows of T in bf16
  __shared__ float  sNorm[32];

  const int tid = threadIdx.x;
  const int r0  = blockIdx.x * 32;

  // Stage 32 rows of T (f32 -> bf16) into LDS. HBM-optimal: one read per row.
  for (int i = tid; i < 32 * DD / 4; i += 256) {
    int off = i * 4;
    int row = off / DD, col = off % DD;     // col is a multiple of 4
    float4 v = make_float4(0.f, 0.f, 0.f, 0.f);
    if (r0 + row < R) v = *(const float4*)(T + (size_t)(r0 + row) * DD + col);
    sA[off + 0] = (__bf16)v.x; sA[off + 1] = (__bf16)v.y;
    sA[off + 2] = (__bf16)v.z; sA[off + 3] = (__bf16)v.w;
  }
  __syncthreads();

  const int wave = tid >> 5, lane = tid & 31;
  const int hf = lane >> 4, ln = lane & 15;

  v8f acc[2][6];
#pragma unroll
  for (int mt = 0; mt < 2; ++mt)
#pragma unroll
    for (int nt = 0; nt < 6; ++nt) acc[mt][nt] = {};

  for (int kk = 0; kk < DD / 32; ++kk) {
    const int k0 = kk * 32;
    v16bf A[2];
#pragma unroll
    for (int mt = 0; mt < 2; ++mt) {
      // lane (hf,ln) holds A[m, hf*16 .. +15] (contiguous, matches B)
      const int m = mt * 16 + ln;
      v8bf lo = *(const v8bf*)&sA[m * DD + k0 + hf * 16];
      v8bf hi = *(const v8bf*)&sA[m * DD + k0 + hf * 16 + 8];
      A[mt] = join16(lo, hi);
    }
#pragma unroll
    for (int nt = 0; nt < 6; ++nt) {
      // B column n = W row n (H = T @ W^T); lane holds W[n, k0+hf*16 .. +15]
      const int n = wave * 96 + nt * 16 + ln;
      const __bf16* wr = Wbf + (size_t)n * DD + k0 + hf * 16;
      v16bf Bm = join16(*(const v8bf*)(wr), *(const v8bf*)(wr + 8));
      acc[0][nt] = wmma_bf16(A[0], Bm, acc[0][nt]);
      acc[1][nt] = wmma_bf16(A[1], Bm, acc[1][nt]);
    }
  }

  if (tid < 32) sNorm[tid] = 0.f;
  __syncthreads();

  // Epilogue: bias + SiLU + residual; accumulate row norms in LDS.
#pragma unroll
  for (int mt = 0; mt < 2; ++mt) {
#pragma unroll
    for (int r = 0; r < 8; ++r) {
      const int rl = mt * 16 + hf * 8 + r;     // C/D layout: M = r + 8*hf
      const int row = r0 + rl;
      float sq = 0.f;
      if (row < R) {
#pragma unroll
        for (int nt = 0; nt < 6; ++nt) {
          const int col = wave * 96 + nt * 16 + ln;
          float h = acc[mt][nt][r] + bias[col];
          float s = h / (1.f + __expf(-h));          // SiLU
          float v = s + T[(size_t)row * DD + col];   // residual (exact f32 t)
          acc[mt][nt][r] = v;
          sq += v * v;
        }
      }
      atomicAdd(&sNorm[rl], sq);
    }
  }
  __syncthreads();

#pragma unroll
  for (int mt = 0; mt < 2; ++mt) {
#pragma unroll
    for (int r = 0; r < 8; ++r) {
      const int rl = mt * 16 + hf * 8 + r;
      const int row = r0 + rl;
      if (row >= R) continue;
      const float inv = 1.f / fmaxf(sqrtf(sNorm[rl]), EPS_);
      const int lab = (mode == 1) ? labels[row] : 0;
#pragma unroll
      for (int nt = 0; nt < 6; ++nt) {
        const int col = wave * 96 + nt * 16 + ln;
        const float v = acc[mt][nt][r] * inv;
        if (mode == 0) out_emb[(size_t)row * DD + col] = v;
        else           atomicAdd(&sums[(size_t)lab * DD + col], v);
      }
    }
  }
  if (mode == 1 && tid < 32 && (r0 + tid) < R)
    atomicAdd(&counts[labels[r0 + tid]], 1.0f);
}

// ---------------------------------------------------------------------------
// proto_kernel: prototypes[c] = count>0 ? l2norm(sums[c]/max(count,1)) : 0
// Writes bf16 prototypes directly (consumed only by the WMMA logits GEMM).
// ---------------------------------------------------------------------------
__global__ __launch_bounds__(256)
void proto_kernel(const float* __restrict__ sums, const float* __restrict__ counts,
                  __bf16* __restrict__ protobf)
{
  __shared__ float red[256];
  const int c = blockIdx.x, tid = threadIdx.x;
  const float cnt = counts[c];
  const float invc = 1.f / fmaxf(cnt, 1.f);
  float m0 = sums[(size_t)c * DD + tid      ] * invc;
  float m1 = sums[(size_t)c * DD + tid + 256] * invc;
  float m2 = sums[(size_t)c * DD + tid + 512] * invc;
  red[tid] = m0 * m0 + m1 * m1 + m2 * m2;
  __syncthreads();
  for (int s = 128; s > 0; s >>= 1) {
    if (tid < s) red[tid] += red[tid + s];
    __syncthreads();
  }
  const float inv = (cnt > 0.f) ? 1.f / fmaxf(sqrtf(red[0]), EPS_) : 0.f;
  protobf[(size_t)c * DD + tid      ] = (__bf16)(m0 * inv);
  protobf[(size_t)c * DD + tid + 256] = (__bf16)(m1 * inv);
  protobf[(size_t)c * DD + tid + 512] = (__bf16)(m2 * inv);
}

// ---------------------------------------------------------------------------
// logits_kernel: logits = exp(ALPHA*(emb @ protos^T - 1)) / TAU
// Block tile 64 rows x 64 cols; 8 waves x (1 mt, 2 nt) covers 16 WMMA tiles.
// Both operands are pre-converted bf16 in L2; fragments are contiguous
// 16-byte chunks -> no LDS staging, no barriers in the K loop.
// ---------------------------------------------------------------------------
__global__ __launch_bounds__(256)
void logits_kernel(const __bf16* __restrict__ embbf, const __bf16* __restrict__ protobf,
                   float* __restrict__ logits)
{
  const int tid = threadIdx.x;
  const int rb0 = blockIdx.x * 64;
  const int cb0 = blockIdx.y * 64;
  const int wave = tid >> 5, lane = tid & 31;
  const int hf = lane >> 4, ln = lane & 15;
  const int t0 = wave * 2;
  const int mt = t0 >> 2;        // row tile for this wave
  const int nt0 = t0 & 3;        // first of two col tiles

  const int arow = rb0 + mt * 16 + ln;
  const __bf16* abase = embbf + (size_t)arow * DD;

  v8f acc[2];
  acc[0] = {}; acc[1] = {};

  const int n0 = cb0 + nt0 * 16 + ln;
  const int n1 = cb0 + (nt0 + 1) * 16 + ln;
  const __bf16* pb0 = protobf + (size_t)n0 * DD;
  const __bf16* pb1 = protobf + (size_t)n1 * DD;

  for (int kk = 0; kk < DD / 32; ++kk) {
    const int k0 = kk * 32;
    v16bf A = join16(*(const v8bf*)(abase + k0 + hf * 16),
                     *(const v8bf*)(abase + k0 + hf * 16 + 8));
    v16bf B0 = (n0 < CC)
      ? join16(*(const v8bf*)(pb0 + k0 + hf * 16), *(const v8bf*)(pb0 + k0 + hf * 16 + 8))
      : zero16bf();
    v16bf B1 = (n1 < CC)
      ? join16(*(const v8bf*)(pb1 + k0 + hf * 16), *(const v8bf*)(pb1 + k0 + hf * 16 + 8))
      : zero16bf();
    acc[0] = wmma_bf16(A, B0, acc[0]);
    acc[1] = wmma_bf16(A, B1, acc[1]);
  }

#pragma unroll
  for (int j = 0; j < 2; ++j) {
#pragma unroll
    for (int r = 0; r < 8; ++r) {
      const int row = rb0 + mt * 16 + hf * 8 + r;
      const int col = cb0 + (nt0 + j) * 16 + ln;
      if (col < CC) {
        const float cosv = acc[j][r];
        logits[(size_t)row * CC + col] = __expf(ALPHA_ * (cosv - 1.f)) * (1.0f / TAU_);
      }
    }
  }
}

// ---------------------------------------------------------------------------
// softmax_kernel: predicts = softmax(logits, axis=-1), one block per row.
// ---------------------------------------------------------------------------
__global__ __launch_bounds__(256)
void softmax_kernel(const float* __restrict__ logits, float* __restrict__ predicts)
{
  __shared__ float red[256];
  const int b = blockIdx.x, tid = threadIdx.x;
  const float* lrow = logits + (size_t)b * CC;

  float mx = -1e30f;
  for (int c = tid; c < CC; c += 256) mx = fmaxf(mx, lrow[c]);
  red[tid] = mx; __syncthreads();
  for (int s = 128; s > 0; s >>= 1) {
    if (tid < s) red[tid] = fmaxf(red[tid], red[tid + s]);
    __syncthreads();
  }
  mx = red[0];
  __syncthreads();

  float sum = 0.f;
  for (int c = tid; c < CC; c += 256) sum += __expf(lrow[c] - mx);
  red[tid] = sum; __syncthreads();
  for (int s = 128; s > 0; s >>= 1) {
    if (tid < s) red[tid] += red[tid + s];
    __syncthreads();
  }
  const float inv = 1.f / red[0];
  for (int c = tid; c < CC; c += 256)
    predicts[(size_t)b * CC + c] = __expf(lrow[c] - mx) * inv;
}

__global__ void zero_kernel(float* __restrict__ p, int n) {
  int i = blockIdx.x * blockDim.x + threadIdx.x;
  if (i < n) p[i] = 0.f;
}

// ---------------------------------------------------------------------------
extern "C" void kernel_launch(void* const* d_in, const int* in_sizes, int n_in,
                              void* d_out, int out_size, void* d_ws, size_t ws_size,
                              hipStream_t stream) {
  const float* x      = (const float*)d_in[0];
  const float* sf     = (const float*)d_in[1];
  const int*   labels = (const int*)d_in[2];
  const float* W      = (const float*)d_in[3];
  const float* bias   = (const float*)d_in[4];

  const int Bn = in_sizes[0] / DD;   // 8192
  const int Nn = in_sizes[2];        // 100000

  // Workspace layout (~18.4 MB):
  //   sums   : CC*DD f32      (3.07 MB)
  //   counts : 1024 f32
  //   Wbf    : DD*DD bf16     (1.18 MB, L2-resident)
  //   protobf: CC*DD bf16     (1.54 MB, L2-resident)
  //   embbf  : Bn*DD bf16     (12.6 MB)
  float*  sums    = (float*)d_ws;
  float*  counts  = sums + (size_t)CC * DD;
  __bf16* Wbf     = (__bf16*)(counts + 1024);
  __bf16* protobf = Wbf + (size_t)DD * DD;
  __bf16* embbf   = protobf + (size_t)CC * DD;

  float* out_pred = (float*)d_out;
  float* out_log  = out_pred + (size_t)Bn * CC;
  float* out_emb  = out_log  + (size_t)Bn * CC;

  const int zn = CC * DD + CC;   // counts are contiguous right after sums
  zero_kernel<<<(zn + 255) / 256, 256, 0, stream>>>(sums, zn);

  // W f32 -> bf16 once (re-used by 3381 adapter blocks from L2).
  {
    const int n4 = DD * DD / 4;
    cvt_bf16_kernel<<<(n4 + 255) / 256, 256, 0, stream>>>(W, Wbf, n4);
  }

  adapter_kernel<<<(Bn + 31) / 32, 256, 0, stream>>>(
      x, Wbf, bias, labels, Bn, out_emb, sums, counts, /*mode=*/0);

  adapter_kernel<<<(Nn + 31) / 32, 256, 0, stream>>>(
      sf, Wbf, bias, labels, Nn, out_emb, sums, counts, /*mode=*/1);

  // Embeddings f32 -> bf16 copy for the logits GEMM (read 16x from L2).
  {
    const int n4 = Bn * DD / 4;
    cvt_bf16_kernel<<<(n4 + 255) / 256, 256, 0, stream>>>(out_emb, embbf, n4);
  }

  proto_kernel<<<CC, 256, 0, stream>>>(sums, counts, protobf);

  dim3 lg((Bn + 63) / 64, (CC + 63) / 64);
  logits_kernel<<<lg, 256, 0, stream>>>(embbf, protobf, out_log);

  softmax_kernel<<<Bn, 256, 0, stream>>>(out_log, out_pred);
}
